// HetBlock_29695403884874
// MI455X (gfx1250) — matrix operand order repaired and checked by reading, not compiled, not run on hardware
//
#include <hip/hip_runtime.h>
#include <hip/hip_bf16.h>

typedef __attribute__((ext_vector_type(16))) __bf16 v16bf;
typedef __attribute__((ext_vector_type(8)))  __bf16 v8bf;
typedef __attribute__((ext_vector_type(8)))  float  v8f;

#define DFEAT 128
#define KDIM  256
#define LN_EPS 1e-5f

// ---------- helpers ----------
static __device__ __forceinline__ unsigned short f32_to_bf16(float f) {
    unsigned u = __float_as_uint(f);
    unsigned r = (u + 0x7FFFu + ((u >> 16) & 1u)) >> 16;   // round-to-nearest-even
    return (unsigned short)r;
}

static __device__ __forceinline__ void atomic_add_f32(float* p, float v) {
    __hip_atomic_fetch_add(p, v, __ATOMIC_RELAXED, __HIP_MEMORY_SCOPE_AGENT);
}

// ---------- zero fill ----------
__global__ void zero_f32(float* p, long long n) {
    long long i = (long long)blockIdx.x * blockDim.x + threadIdx.x;
    long long stride = (long long)gridDim.x * blockDim.x;
    for (; i < n; i += stride) p[i] = 0.0f;
}

// ---------- edge scatter: sum[dst] += h_src[src], cnt[dst] += 1 ----------
// one wave per edge; each lane handles 4 contiguous floats
__global__ void scatter_sum(const float* __restrict__ hs,
                            const int* __restrict__ src,
                            const int* __restrict__ dst,
                            float* __restrict__ sum,
                            float* __restrict__ cnt, int E) {
    int t = blockIdx.x * blockDim.x + threadIdx.x;
    int e = t >> 5;
    int lane = t & 31;
    if (e >= E) return;
    int s = src[e];
    int d = dst[e];
    const float4 v = *(const float4*)(hs + (size_t)s * DFEAT + lane * 4);
    float* base = sum + (size_t)d * DFEAT + lane * 4;
    atomic_add_f32(base + 0, v.x);
    atomic_add_f32(base + 1, v.y);
    atomic_add_f32(base + 2, v.z);
    atomic_add_f32(base + 3, v.w);
    if (lane == 0) atomic_add_f32(cnt + d, 1.0f);
}

// ---------- pack X = [x_dst | sum/max(cnt,1)] as bf16 [N,256] ----------
__global__ void pack_x(const float* __restrict__ xd,
                       const float* __restrict__ sum,
                       const float* __restrict__ cnt,
                       unsigned short* __restrict__ X, int N) {
    int t = blockIdx.x * blockDim.x + threadIdx.x;
    int row = t >> 5;
    int lane = t & 31;
    if (row >= N) return;
    float c = cnt[row];
    float inv = 1.0f / fmaxf(c, 1.0f);
    unsigned short* xr = X + (size_t)row * KDIM;
    const float* xdr = xd + (size_t)row * DFEAT;
    const float* smr = sum + (size_t)row * DFEAT;
#pragma unroll
    for (int i = 0; i < 4; ++i) {
        int col = lane + i * 32;
        xr[col]         = f32_to_bf16(xdr[col]);
        xr[DFEAT + col] = f32_to_bf16(smr[col] * inv);
    }
}

// ---------- convert W [128,256] f32 -> bf16 ----------
__global__ void conv_w(const float* __restrict__ W, unsigned short* __restrict__ Wbf) {
    int i = blockIdx.x * blockDim.x + threadIdx.x;
    if (i < DFEAT * KDIM) Wbf[i] = f32_to_bf16(W[i]);
}

// ---------- WMMA GEMM: out[n,j] += sum_k X[n,k]*W[j,k] + bias[j] ----------
// block = 256 threads = 8 waves; blockIdx.x = M-tile (16 rows), wave id = N-tile (16 cols)
__global__ void gemm_wmma_bf16(const unsigned short* __restrict__ Xu,
                               const unsigned short* __restrict__ Wu,
                               const float* __restrict__ bias,
                               float* __restrict__ out) {
    const int wave  = threadIdx.x >> 5;   // 0..7 -> N-tile
    const int lane  = threadIdx.x & 31;
    const int l15   = lane & 15;
    const int khalf = lane >> 4;          // 0 or 1
    const int row0  = blockIdx.x * 16;

    const __bf16* Xr = ((const __bf16*)Xu) + (size_t)(row0 + l15) * KDIM;
    const __bf16* Wr = ((const __bf16*)Wu) + (size_t)(wave * 16 + l15) * KDIM;

    v8f c = {};
#pragma unroll
    for (int kt = 0; kt < 8; ++kt) {
        const int k0 = kt * 32;
        // A fragment: lanes 0-15 K k0+0..15; lanes 16-31 K k0+16..31
        v8bf alo = *(const v8bf*)(Xr + k0 + khalf * 16);
        v8bf ahi = *(const v8bf*)(Xr + k0 + khalf * 16 + 8);
        v16bf a;
#pragma unroll
        for (int i = 0; i < 8; ++i) { a[i] = alo[i]; a[8 + i] = ahi[i]; }
        // B fragment: lanes 0-15 K k0+0..15, lanes 16-31 K k0+16..31 (contiguous in W row)
        v16bf b = *(const v16bf*)(Wr + k0 + khalf * 16);
        c = __builtin_amdgcn_wmma_f32_16x16x32_bf16(false, a, false, b,
                                                    (short)0, c, false, false);
    }

    const int col = wave * 16 + l15;
    const float bv = bias[col];
    float* orow = out + (size_t)(row0 + khalf * 8) * DFEAT + col;
#pragma unroll
    for (int r = 0; r < 8; ++r) {
        float* p = orow + (size_t)r * DFEAT;
        *p = *p + c[r] + bv;
    }
}

// ---------- LayerNorm (population var, eps=1e-5), one wave per row ----------
__global__ void layernorm_rows(float* __restrict__ out,
                               const float* __restrict__ g,
                               const float* __restrict__ be, int N) {
    int t = blockIdx.x * blockDim.x + threadIdx.x;
    int row = t >> 5;
    int lane = t & 31;
    if (row >= N) return;
    float* r = out + (size_t)row * DFEAT;
    float v[4];
    float s = 0.0f, s2 = 0.0f;
#pragma unroll
    for (int i = 0; i < 4; ++i) {
        v[i] = r[lane + i * 32];
        s += v[i];
        s2 += v[i] * v[i];
    }
#pragma unroll
    for (int off = 16; off >= 1; off >>= 1) {
        s  += __shfl_xor(s,  off, 32);
        s2 += __shfl_xor(s2, off, 32);
    }
    const float mu  = s * (1.0f / DFEAT);
    const float var = s2 * (1.0f / DFEAT) - mu * mu;
    const float rs  = rsqrtf(var + LN_EPS);
#pragma unroll
    for (int i = 0; i < 4; ++i) {
        int col = lane + i * 32;
        r[col] = (v[i] - mu) * rs * g[col] + be[col];
    }
}

// ---------- host launch ----------
extern "C" void kernel_launch(void* const* d_in, const int* in_sizes, int n_in,
                              void* d_out, int out_size, void* d_ws, size_t ws_size,
                              hipStream_t stream) {
    const float* h_a = (const float*)d_in[0];
    const float* h_b = (const float*)d_in[1];
    const int* ei_ab = (const int*)d_in[2];
    const int* ei_ba = (const int*)d_in[3];
    const int* ei_aa = (const int*)d_in[4];
    const float* W_ab = (const float*)d_in[5];
    const float* b_ab = (const float*)d_in[6];
    const float* W_ba = (const float*)d_in[7];
    const float* b_ba = (const float*)d_in[8];
    const float* W_aa = (const float*)d_in[9];
    const float* b_aa = (const float*)d_in[10];
    const float* g_a  = (const float*)d_in[11];
    const float* be_a = (const float*)d_in[12];
    const float* g_b  = (const float*)d_in[13];
    const float* be_b = (const float*)d_in[14];

    const int N_A = in_sizes[0] / DFEAT;
    const int N_B = in_sizes[1] / DFEAT;
    const int E   = in_sizes[2] / 2;
    const int Nmax = (N_A > N_B) ? N_A : N_B;

    float* out_a = (float*)d_out;
    float* out_b = out_a + (size_t)N_A * DFEAT;

    // workspace carve-up
    char* ws = (char*)d_ws;
    size_t sum_bytes = (size_t)Nmax * DFEAT * sizeof(float);
    size_t cnt_bytes = (size_t)Nmax * sizeof(float);
    float* sum = (float*)ws;                      ws += (sum_bytes + 255) & ~(size_t)255;
    float* cnt = (float*)ws;                      ws += (cnt_bytes + 255) & ~(size_t)255;
    unsigned short* X   = (unsigned short*)ws;    ws += (((size_t)Nmax * KDIM * 2) + 255) & ~(size_t)255;
    unsigned short* Wbf = (unsigned short*)ws;

    const int TB = 256;
    // zero the full output (out_a accumulates two edge types)
    {
        long long n = (long long)(N_A + N_B) * DFEAT;
        int blocks = 4096;
        zero_f32<<<blocks, TB, 0, stream>>>((float*)d_out, n);
    }

    struct Et { const float* hs; const float* xd; const int* ei; const float* W;
                const float* b; float* out; int Nd; };
    Et ets[3] = {
        { h_a, h_b, ei_ab, W_ab, b_ab, out_b, N_B },  // ('a','to','b')
        { h_b, h_a, ei_ba, W_ba, b_ba, out_a, N_A },  // ('b','to','a')
        { h_a, h_a, ei_aa, W_aa, b_aa, out_a, N_A },  // ('a','self','a')
    };

    for (int t = 0; t < 3; ++t) {
        const Et& et = ets[t];
        // zero sums/counts
        zero_f32<<<4096, TB, 0, stream>>>(sum, (long long)et.Nd * DFEAT);
        zero_f32<<<256, TB, 0, stream>>>(cnt, (long long)et.Nd);
        // scatter: one wave per edge
        {
            long long threads = (long long)E * 32;
            int blocks = (int)((threads + TB - 1) / TB);
            scatter_sum<<<blocks, TB, 0, stream>>>(et.hs, et.ei, et.ei + E, sum, cnt, E);
        }
        // pack X = [x_dst | agg] bf16
        {
            long long threads = (long long)et.Nd * 32;
            int blocks = (int)((threads + TB - 1) / TB);
            pack_x<<<blocks, TB, 0, stream>>>(et.xd, sum, cnt, X, et.Nd);
        }
        // W -> bf16
        conv_w<<<(DFEAT * KDIM + TB - 1) / TB, TB, 0, stream>>>(et.W, Wbf);
        // WMMA GEMM (+bias) accumulate into out
        {
            int mtiles = et.Nd / 16;   // N = 100000 divisible by 16
            gemm_wmma_bf16<<<mtiles, TB, 0, stream>>>(X, Wbf, et.b, et.out);
        }
    }

    // layer norms
    {
        long long ta = (long long)N_A * 32;
        layernorm_rows<<<(int)((ta + TB - 1) / TB), TB, 0, stream>>>(out_a, g_a, be_a, N_A);
        long long tb = (long long)N_B * 32;
        layernorm_rows<<<(int)((tb + TB - 1) / TB), TB, 0, stream>>>(out_b, g_b, be_b, N_B);
    }
}
